// PerSpanAttention_292057776140
// MI455X (gfx1250) — matrix operand order repaired and checked by reading, not compiled, not run on hardware
//
#include <hip/hip_runtime.h>

// ---------------------------------------------------------------------------
// PerSpanAttention on MI455X (gfx1250, wave32, WMMA)
//   1) f32 -> bf16 conversions for q, Wq, Wkv, Wo (workspace)
//   2) qh = q @ Wq^T + bq           (WMMA bf16 GEMM, bf16 out)
//   3) fused per-span: kvh proj (WMMA from LDS-staged kv) + softmax + P@V
//      -- two heads per iteration: 8 accumulators/wave, 1.5 b128 loads/WMMA
//   4) out = attn @ Wo^T + bo       (WMMA bf16 GEMM, f32 out)
// WMMA loops software-pipelined (ping-pong buffers, peeled tail); global
// fragment loads use uniform-base + 32-bit lane offsets so the backend can
// use SADDR+VGPR32 addressing instead of per-lane 64-bit pointer adds.
// ---------------------------------------------------------------------------

typedef __attribute__((ext_vector_type(16))) __bf16 v16bf;
typedef __attribute__((ext_vector_type(8)))  float  v8f;

#define BSZ        8
#define LSZ        8192
#define HSIZE      1024
#define NHEADS     16
#define HDIM       64
#define SPANSZ     64
#define NSPAN      128            // LSZ / SPANSZ
#define TOTSPAN    1024           // BSZ * NSPAN

#define KV_LDS_STRIDE 1032        // (1024 + 8) bf16 -> conflict-free A-frag reads
#define KHV_STRIDE    72          // (64 + 8) bf16

union Frag16 {                    // 16 bf16 values = one WMMA A/B fragment per lane
  v16bf v;
  unsigned short u[16];
  uint4 q[2];
};

static __device__ __forceinline__ unsigned short f2bf(float f) {
  unsigned u = __float_as_uint(f);
  unsigned r = 0x7FFFu + ((u >> 16) & 1u);     // round-to-nearest-even
  return (unsigned short)((u + r) >> 16);
}
static __device__ __forceinline__ float bf2f(unsigned short s) {
  return __uint_as_float(((unsigned)s) << 16);
}
static __device__ __forceinline__ unsigned pack2bf(float lo, float hi) {
  return (unsigned)f2bf(lo) | ((unsigned)f2bf(hi) << 16);
}

#define WMMA_BF16(A, Bf, C) \
  __builtin_amdgcn_wmma_f32_16x16x32_bf16(false, (A).v, false, (Bf).v, (short)0, (C), false, false)

// ---------------------------------------------------------------------------
// f32 -> bf16 conversion, vectorized (n must be a multiple of 4)
// ---------------------------------------------------------------------------
__global__ void cvt_f32_bf16(const float* __restrict__ in,
                             unsigned short* __restrict__ out, int n) {
  const int n4 = n >> 2;
  int i = blockIdx.x * blockDim.x + threadIdx.x;
  const int stride = gridDim.x * blockDim.x;
  const float4* in4 = (const float4*)in;
  uint2* out2 = (uint2*)out;
  for (; i < n4; i += stride) {
    float4 f = in4[i];
    uint2 o;
    o.x = pack2bf(f.x, f.y);
    o.y = pack2bf(f.z, f.w);
    out2[i] = o;
  }
}

// ---------------------------------------------------------------------------
// C(1024,1024) = A(1024,1024) @ B(1024,1024)^T + bias (bf16 in; bf16/f32 out)
// Each wave: 32(M) x 64(N) strip = 2 A-frags x 4 B-frags x 8 accumulators.
// K-loop unrolled by 2 with ping-pong buffers; tail peeled (no wrap selects).
// ---------------------------------------------------------------------------
__global__ __launch_bounds__(256) void gemm_bf16_wmma(
    const unsigned short* __restrict__ A, const unsigned short* __restrict__ Bw,
    const float* __restrict__ bias, unsigned short* __restrict__ outB,
    float* __restrict__ outF, int writeF32) {
  const int lane = threadIdx.x & 31;
  const int wave = threadIdx.x >> 5;
  const int gw = blockIdx.x * 8 + wave;
  const int nStrips = HSIZE >> 6;              // 16
  const int mrow = (gw / nStrips) << 5;        // 32-row strip
  const int ncol = (gw % nStrips) << 6;        // 64-col strip
  const int half = lane >> 4;
  const int l16  = lane & 15;

  v8f acc[2][4] = {{{}, {}, {}, {}}, {{}, {}, {}, {}}};

  // 32-bit element offsets from uniform bases (enables SADDR+VGPR32 loads)
  unsigned aOff[2], bOff[4];
#pragma unroll
  for (int i = 0; i < 2; ++i)
    aOff[i] = (unsigned)((mrow + i * 16 + l16) * HSIZE + half * 16);
#pragma unroll
  for (int nt = 0; nt < 4; ++nt)
    bOff[nt] = (unsigned)((ncol + nt * 16 + l16) * HSIZE + half * 16);

  Frag16 a0[2], b0[4], a1[2], b1[4];

  auto loadStep = [&](Frag16* a, Frag16* b, int kk) {
#pragma unroll
    for (int i = 0; i < 2; ++i) {
      a[i].q[0] = *(const uint4*)(A + aOff[i] + kk);
      a[i].q[1] = *(const uint4*)(A + aOff[i] + kk + 8);
    }
#pragma unroll
    for (int nt = 0; nt < 4; ++nt) {
      b[nt].q[0] = *(const uint4*)(Bw + bOff[nt] + kk);
      b[nt].q[1] = *(const uint4*)(Bw + bOff[nt] + kk + 8);
    }
  };
  auto mmaStep = [&](Frag16* a, Frag16* b) {
#pragma unroll
    for (int i = 0; i < 2; ++i)
#pragma unroll
      for (int nt = 0; nt < 4; ++nt)
        acc[i][nt] = WMMA_BF16(a[i], b[nt], acc[i][nt]);
  };

  loadStep(a0, b0, 0);
  int kk = 0;
  for (; kk < HSIZE - 64; kk += 64) {
    loadStep(a1, b1, kk + 32);                 // prefetch k+32
    mmaStep(a0, b0);                           // compute k
    loadStep(a0, b0, kk + 64);                 // prefetch k+64
    mmaStep(a1, b1);                           // compute k+32
  }
  loadStep(a1, b1, kk + 32);                   // peeled tail
  mmaStep(a0, b0);
  mmaStep(a1, b1);

#pragma unroll
  for (int i = 0; i < 2; ++i) {
#pragma unroll
    for (int nt = 0; nt < 4; ++nt) {
      const int c = ncol + nt * 16 + l16;
      const float bb = bias[c];
#pragma unroll
      for (int v = 0; v < 8; ++v) {
        const int r = mrow + i * 16 + v + half * 8;  // C/D layout: M = v + 8*half
        const float val = acc[i][nt][v] + bb;
        if (writeF32) outF[(size_t)r * HSIZE + c] = val;
        else          outB[(size_t)r * HSIZE + c] = f2bf(val);
      }
    }
  }
}

// ---------------------------------------------------------------------------
// Fused per-span kernel. One workgroup (8 waves) per (b, span).
//   LDS: kv slice (64x1024 bf16) + qh row + K/V for 2 heads + softmax scratch
//   Per head-pair: wave (mat = K|V, nt = d-tile) computes 2 heads x 4 m-tiles
//   = 8 accumulators; per k-step: 8 ds_load_b128 + 4 global_load_b128 feed
//   8 v_wmma (1.5 loads/WMMA). Ping-pong software pipeline, peeled tail.
// ---------------------------------------------------------------------------
__global__ __launch_bounds__(256) void span_attn_wmma(
    const float* __restrict__ kv, const unsigned short* __restrict__ WkvB,
    const float* __restrict__ bkv, const unsigned short* __restrict__ qhB,
    const unsigned char* __restrict__ mask, unsigned short* __restrict__ attnB) {
  extern __shared__ char smem[];
  unsigned short* kvL  = (unsigned short*)smem;                    // 64 x 1032
  unsigned short* qrow = kvL + 64 * KV_LDS_STRIDE;                 // 1024
  unsigned short* khv  = qrow + HSIZE;                             // 4 x 64 x 72
  float* sc  = (float*)(khv + 4 * SPANSZ * KHV_STRIDE);            // 2 x 64
  float* red = sc + 2 * SPANSZ;                                    // scratch

  const int span = blockIdx.x;                 // = b*NSPAN + n
  const int tid  = threadIdx.x;
  const int lane = tid & 31;
  const int wave = tid >> 5;
  const int half = lane >> 4;
  const int l16  = lane & 15;
  const int mat  = wave >> 2;                  // 0 = K, 1 = V
  const int nt   = wave & 3;                   // d-tile (16 cols of head dim)

  // ---- stage 1: kv span (64 x 1024 f32) -> bf16 LDS, plus qh row ----
  const float* kvSpan = kv + (size_t)span * SPANSZ * HSIZE;
  for (int i = tid; i < (SPANSZ * HSIZE) / 4; i += 256) {
    const int row  = i >> 8;                   // 256 float4 per row
    const int col4 = i & 255;
    const float4 f = ((const float4*)(kvSpan + (size_t)row * HSIZE))[col4];
    uint2* dst = (uint2*)(kvL + row * KV_LDS_STRIDE + col4 * 4);
    uint2 o;
    o.x = pack2bf(f.x, f.y);
    o.y = pack2bf(f.z, f.w);
    *dst = o;
  }
  for (int i = tid; i < HSIZE; i += 256) qrow[i] = qhB[(size_t)span * HSIZE + i];
  __syncthreads();

  // 32-bit element offset from uniform WkvB base (SADDR+VGPR32 addressing)
  const unsigned bOffBase =
      (unsigned)((mat * HSIZE + nt * 16 + l16) * HSIZE + half * 16);
  const unsigned short* aBase = kvL + l16 * KV_LDS_STRIDE + half * 16;

  for (int hp = 0; hp < NHEADS / 2; ++hp) {
    unsigned bOff[2];
    bOff[0] = bOffBase + (unsigned)((hp * 2 + 0) * HDIM * HSIZE);
    bOff[1] = bOffBase + (unsigned)((hp * 2 + 1) * HDIM * HSIZE);

    v8f acc[2][4] = {{{}, {}, {}, {}}, {{}, {}, {}, {}}};
    Frag16 a0[4], a1[4], b0[2], b1[2];

    auto loadStep = [&](Frag16* a, Frag16* b, int kk) {
#pragma unroll
      for (int s = 0; s < 2; ++s) {
        b[s].q[0] = *(const uint4*)(WkvB + bOff[s] + kk);
        b[s].q[1] = *(const uint4*)(WkvB + bOff[s] + kk + 8);
      }
#pragma unroll
      for (int mt = 0; mt < 4; ++mt) {
        const unsigned short* ar = aBase + mt * 16 * KV_LDS_STRIDE + kk;
        a[mt].q[0] = *(const uint4*)(ar);
        a[mt].q[1] = *(const uint4*)(ar + 8);
      }
    };
    auto mmaStep = [&](Frag16* a, Frag16* b) {
#pragma unroll
      for (int s = 0; s < 2; ++s)
#pragma unroll
        for (int mt = 0; mt < 4; ++mt)
          acc[s][mt] = WMMA_BF16(a[mt], b[s], acc[s][mt]);
    };

    loadStep(a0, b0, 0);
    int kk = 0;
    for (; kk < HSIZE - 64; kk += 64) {
      loadStep(a1, b1, kk + 32);               // prefetch k+32
      mmaStep(a0, b0);                         // compute k
      loadStep(a0, b0, kk + 64);               // prefetch k+64
      mmaStep(a1, b1);                         // compute k+32
    }
    loadStep(a1, b1, kk + 32);                 // peeled tail
    mmaStep(a0, b0);
    mmaStep(a1, b1);

    // bias + store K/V strips for both heads to LDS as bf16
#pragma unroll
    for (int s = 0; s < 2; ++s) {
      const float bias = bkv[mat * HSIZE + (hp * 2 + s) * HDIM + nt * 16 + l16];
#pragma unroll
      for (int mt = 0; mt < 4; ++mt) {
#pragma unroll
        for (int v = 0; v < 8; ++v) {
          const int sr = mt * 16 + v + half * 8;            // span row
          const int d  = nt * 16 + l16;
          khv[((s * 2 + mat) * SPANSZ + sr) * KHV_STRIDE + d] =
              f2bf(acc[s][mt][v] + bias);
        }
      }
    }
    __syncthreads();

    // ---- scores: 128 lanes = 2 heads x 64 span rows ----
    if (tid < 2 * SPANSZ) {
      const int hh = tid >> 6;                              // head within pair
      const int sr = tid & 63;
      float a = 0.f;
#pragma unroll 4
      for (int d = 0; d < HDIM; ++d)
        a += bf2f(qrow[(hp * 2 + hh) * HDIM + d]) *
             bf2f(khv[((hh * 2 + 0) * SPANSZ + sr) * KHV_STRIDE + d]);
      const bool mk = mask[(size_t)span * SPANSZ + sr] != 0;
      sc[hh * SPANSZ + sr] = mk ? a * 0.125f : -3.4e38f;    // scale = 1/sqrt(64)
    }
    __syncthreads();

    // ---- softmax over 64, one serial thread per head (trivially cheap) ----
    if (tid < 2 * SPANSZ && (tid & 63) == 0) {
      const int hh = tid >> 6;
      float* scp = sc + hh * SPANSZ;
      float mx = scp[0];
      for (int s = 1; s < SPANSZ; ++s) mx = fmaxf(mx, scp[s]);
      float sum = 0.f;
      for (int s = 0; s < SPANSZ; ++s) { float e = __expf(scp[s] - mx); scp[s] = e; sum += e; }
      red[hh] = 1.0f / sum;
    }
    __syncthreads();

    // ---- attn_h[d] = sum_s p[s] * V_h[s,d] : 128 lanes = 2 heads x 64 d ----
    if (tid < 2 * SPANSZ) {
      const int hh = tid >> 6;
      const int d  = tid & 63;
      const float inv = red[hh];
      float a = 0.f;
#pragma unroll 4
      for (int s = 0; s < SPANSZ; ++s)
        a += sc[hh * SPANSZ + s] *
             bf2f(khv[((hh * 2 + 1) * SPANSZ + s) * KHV_STRIDE + d]);
      attnB[(size_t)span * HSIZE + (hp * 2 + hh) * HDIM + d] = f2bf(a * inv);
    }
    __syncthreads();
  }
}

// ---------------------------------------------------------------------------
extern "C" void kernel_launch(void* const* d_in, const int* in_sizes, int n_in,
                              void* d_out, int out_size, void* d_ws, size_t ws_size,
                              hipStream_t stream) {
  (void)in_sizes; (void)n_in; (void)out_size; (void)ws_size;
  const float* q            = (const float*)d_in[0];   // (B, N, HS)
  const float* kv           = (const float*)d_in[1];   // (B, L, HS)
  const unsigned char* mask = (const unsigned char*)d_in[2];  // (B, N, SPAN) bool
  // d_in[3] = span_size (compile-time 64)
  const float* Wq  = (const float*)d_in[4];            // (1024, 1024)
  const float* bq  = (const float*)d_in[5];
  const float* Wkv = (const float*)d_in[6];            // (2048, 1024)
  const float* bkv = (const float*)d_in[7];
  const float* Wo  = (const float*)d_in[8];            // (1024, 1024)
  const float* bo  = (const float*)d_in[9];
  float* out = (float*)d_out;

  // workspace layout (bf16 = ushort), total 14 MB
  unsigned short* qB    = (unsigned short*)d_ws;       // 1M
  unsigned short* WqB   = qB    + (1u << 20);          // 1M
  unsigned short* WkvB  = WqB   + (1u << 20);          // 2M
  unsigned short* WoB   = WkvB  + (2u << 20);          // 1M
  unsigned short* qhB   = WoB   + (1u << 20);          // 1M
  unsigned short* attnB = qhB   + (1u << 20);          // 1M

  // 1) conversions
  cvt_f32_bf16<<<512, 256, 0, stream>>>(q,   qB,   1 << 20);
  cvt_f32_bf16<<<512, 256, 0, stream>>>(Wq,  WqB,  1 << 20);
  cvt_f32_bf16<<<1024, 256, 0, stream>>>(Wkv, WkvB, 2 << 20);
  cvt_f32_bf16<<<512, 256, 0, stream>>>(Wo,  WoB,  1 << 20);

  // 2) qh = q @ Wq^T + bq   (1024 x 1024 x 1024) -> bf16
  //    waves = (1024/32) * (1024/64) = 512 -> 64 blocks of 8 waves
  gemm_bf16_wmma<<<64, 256, 0, stream>>>(qB, WqB, bq, qhB, nullptr, 0);

  // 3) fused per-span KV projection + attention
  const size_t smemBytes = (size_t)64 * KV_LDS_STRIDE * 2       // kv slice
                         + (size_t)HSIZE * 2                    // qh row
                         + (size_t)4 * SPANSZ * KHV_STRIDE * 2  // K/V x 2 heads
                         + (2 * SPANSZ + 16) * sizeof(float);   // softmax scratch
  span_attn_wmma<<<TOTSPAN, 256, smemBytes, stream>>>(kv, WkvB, bkv, qhB,
                                                      mask, attnB);

  // 4) out = attn @ Wo^T + bo -> f32
  gemm_bf16_wmma<<<64, 256, 0, stream>>>(attnB, WoB, bo, nullptr, out, 1);
}
